// MultiheadPropagationNet_71811853189807
// MI455X (gfx1250) — matrix operand order repaired and checked by reading, not compiled, not run on hardware
//
#include <hip/hip_runtime.h>
#include <hip/hip_bf16.h>

typedef _Float16 v16h __attribute__((ext_vector_type(16)));
typedef float    v8f  __attribute__((ext_vector_type(8)));

#define NN        4096
#define MWORDS    128          // 4096/32 mask words per row
#define THRESH    0.995f
#define KEEP      0.9f
#define SLOPE     0.2f

// ---------------------------------------------------------------------------
// gfx1250 async copy: global -> LDS, per-lane 16B, tracked by ASYNCcnt.
// GVS addressing: mem = SADDR(64b sgpr pair) + VADDR(32b byte offset).
// ---------------------------------------------------------------------------
__device__ __forceinline__ void async_g2lds_b128(unsigned lds_off, unsigned byte_off,
                                                 const void* sbase) {
    asm volatile("global_load_async_to_lds_b128 %0, %1, %2"
                 :: "v"(lds_off), "v"(byte_off), "s"(sbase)
                 : "memory");
}
__device__ __forceinline__ void wait_async0() {
    asm volatile("s_wait_asynccnt 0" ::: "memory");
}

// ---------------------------------------------------------------------------
// 1) adj (fp32) -> bitmask (adj > THRESH) | eye.  One word = 32 columns.
// ---------------------------------------------------------------------------
__global__ void build_mask(const float* __restrict__ adj, unsigned* __restrict__ mask) {
    int w = blockIdx.x * blockDim.x + threadIdx.x;          // word index
    if (w >= NN * MWORDS) return;
    int row = w / MWORDS;
    int j0  = (w % MWORDS) * 32;
    const float* p = adj + (size_t)row * NN + j0;
    unsigned bits = 0u;
#pragma unroll
    for (int b = 0; b < 32; b += 4) {
        float4 v = *(const float4*)(p + b);
        if (v.x > THRESH) bits |= 1u << (b + 0);
        if (v.y > THRESH) bits |= 1u << (b + 1);
        if (v.z > THRESH) bits |= 1u << (b + 2);
        if (v.w > THRESH) bits |= 1u << (b + 3);
    }
    unsigned d = (unsigned)(row - j0);
    if (d < 32u) bits |= 1u << d;                           // self loop
    mask[w] = bits;
}

// ---------------------------------------------------------------------------
// 2) Generic WMMA GEMM: C[M,N] = A[M,K] @ B[K,N], fp32 in, f16 WMMA, fp32 out.
//    One wave per 16x16 C tile. grid = (M/16, N/16, Z batches).
// ---------------------------------------------------------------------------
__global__ void gemm_wmma(const float* __restrict__ A, const float* __restrict__ B,
                          float* __restrict__ C, int K, int N,
                          long bStrideZ, long cStrideZ) {
    const float* Bz = B + bStrideZ * (long)blockIdx.z;
    float*       Cz = C + cStrideZ * (long)blockIdx.z;
    int m0 = blockIdx.x * 16, n0 = blockIdx.y * 16;
    int lane = threadIdx.x & 31;
    int half = lane >> 4, l = lane & 15;
    v8f c = {};
    int arow = m0 + l;
    for (int k0 = 0; k0 < K; k0 += 32) {
        v16h a, b;
#pragma unroll
        for (int v = 0; v < 8; ++v) {
            int kk = k0 + 2 * v + (half ? 16 : 0);
            float2 x = *(const float2*)(A + (size_t)arow * K + kk);
            a[2 * v]     = (_Float16)x.x;
            a[2 * v + 1] = (_Float16)x.y;
        }
        int klo = k0 + (half ? 16 : 0);
#pragma unroll
        for (int v = 0; v < 8; ++v) {
            b[2 * v]     = (_Float16)Bz[(size_t)(klo + 2 * v) * N + n0 + l];
            b[2 * v + 1] = (_Float16)Bz[(size_t)(klo + 2 * v + 1) * N + n0 + l];
        }
        c = __builtin_amdgcn_wmma_f32_16x16x32_f16(false, a, false, b, (short)0, c,
                                                   false, false);
    }
#pragma unroll
    for (int r = 0; r < 8; ++r) {
        int row = m0 + r + (half ? 8 : 0);
        Cz[(size_t)row * N + n0 + l] = c[r];
    }
}

// ---------------------------------------------------------------------------
// 3) s1[i] = Wh[i,:]·a1, s2[i] = Wh[i,:]·a2.  One wave per row, grid.y = head.
// ---------------------------------------------------------------------------
__global__ void rowdot2(const float* __restrict__ Wh, const float* __restrict__ a1,
                        const float* __restrict__ a2, float* __restrict__ s1,
                        float* __restrict__ s2, int D, long whStrideZ, int aStride,
                        int sStride) {
    int row = blockIdx.x, h = blockIdx.y;
    const float* w  = Wh + whStrideZ * (long)h + (size_t)row * D;
    const float* A1 = a1 + (long)aStride * h;
    const float* A2 = a2 + (long)aStride * h;
    float x1 = 0.f, x2 = 0.f;
    for (int d = threadIdx.x; d < D; d += 32) {
        float v = w[d];
        x1 += v * A1[d];
        x2 += v * A2[d];
    }
#pragma unroll
    for (int o = 16; o; o >>= 1) {
        x1 += __shfl_down(x1, o, 32);
        x2 += __shfl_down(x2, o, 32);
    }
    if (threadIdx.x == 0) {
        s1[(long)sStride * h + row] = x1;
        s2[(long)sStride * h + row] = x2;
    }
}

// ---------------------------------------------------------------------------
// 4) WhT[d][n] (f16) = Wh[n][d] (fp32): B-operand prep for WMMA (contiguous K).
// ---------------------------------------------------------------------------
__global__ void cvt_t(const float* __restrict__ Wh, _Float16* __restrict__ WhT, int D,
                      long whStrideZ, long whtStrideZ) {
    int h   = blockIdx.y;
    int idx = blockIdx.x * blockDim.x + threadIdx.x;        // over NN*D
    if (idx >= NN * D) return;
    int row = idx / D, d = idx % D;
    WhT[whtStrideZ * (long)h + (size_t)d * NN + row] =
        (_Float16)Wh[whStrideZ * (long)h + idx];
}

// ---------------------------------------------------------------------------
// 5) Pass 1: per-row softmax stats (max, sum) of masked leaky-relu logits,
//    fused over H heads from one mask read.  One wave per row.
// ---------------------------------------------------------------------------
template <int H>
__global__ void softmax_stats(const unsigned* __restrict__ mask,
                              const float* __restrict__ s1,
                              const float* __restrict__ s2,
                              float* __restrict__ rmax, float* __restrict__ rsum,
                              int sStride) {
    int wavesPerBlock = blockDim.x >> 5;
    int wave = threadIdx.x >> 5, lane = threadIdx.x & 31;
    int row = blockIdx.x * wavesPerBlock + wave;
    if (row >= NN) return;
    float m[H], ss[H], s1v[H];
#pragma unroll
    for (int h = 0; h < H; ++h) {
        m[h] = -1e30f; ss[h] = 0.f;
        s1v[h] = s1[(long)sStride * h + row];
    }
    const unsigned* mrow = mask + (size_t)row * MWORDS;
    for (int it = 0; it < MWORDS; ++it) {
        unsigned wbits = mrow[it];
        if (!((wbits >> lane) & 1u)) continue;
        int j = it * 32 + lane;
#pragma unroll
        for (int h = 0; h < H; ++h) {
            float e = s1v[h] + s2[(long)sStride * h + j];
            e = e > 0.f ? e : SLOPE * e;
            float nm = fmaxf(m[h], e);
            ss[h] = ss[h] * __expf(m[h] - nm) + __expf(e - nm);
            m[h] = nm;
        }
    }
#pragma unroll
    for (int h = 0; h < H; ++h) {
        float mm = m[h], s = ss[h];
#pragma unroll
        for (int o = 16; o; o >>= 1) {
            float mo = __shfl_xor(mm, o, 32);
            float so = __shfl_xor(s, o, 32);
            float nm = fmaxf(mm, mo);
            s = s * __expf(mm - nm) + so * __expf(mo - nm);
            mm = nm;
        }
        if (lane == 0) {
            rmax[(long)sStride * h + row] = mm;
            rsum[(long)sStride * h + row] = s;
        }
    }
}

// ---------------------------------------------------------------------------
// 6) Pass 2: O = softmax(e) @ Wh via WMMA; res = KEEP*O + (1-KEEP)*Wh (+ELU).
//    8 waves / workgroup, each wave owns 16 rows; the 32-wide K-chunk of the
//    B operand (WhT) is staged into LDS with async global->LDS copies,
//    double-buffered so the DMA overlaps the exp-heavy P-tile build.
//    grid = (NN/128, H).  P tile built directly in the f16 A-operand layout.
// ---------------------------------------------------------------------------
template <int NT, bool ELU>
__global__ void attn_apply(const unsigned* __restrict__ mask,
                           const float* __restrict__ s1, const float* __restrict__ s2,
                           const float* __restrict__ rmax,
                           const float* __restrict__ rsum,
                           const _Float16* __restrict__ WhT,   // [NT*16][NN] f16
                           const float* __restrict__ Wh,       // [NN][NT*16] f32
                           float* __restrict__ out, int outLd, int sStride,
                           long whtStrideZ, long whStrideZ) {
    const int D = NT * 16;
    __shared__ __align__(16) _Float16 smem[2 * NT * 16 * 32];   // [2][D][32]

    int tid  = threadIdx.x;
    int lane = tid & 31;
    int wave = tid >> 5;
    int half = lane >> 4, l = lane & 15;
    int m0 = (blockIdx.x * 8 + wave) * 16;
    int h  = blockIdx.y;
    s1 += (long)sStride * h;  s2 += (long)sStride * h;
    rmax += (long)sStride * h; rsum += (long)sStride * h;
    WhT += whtStrideZ * (long)h;
    Wh  += whStrideZ * (long)h;

    int   row  = m0 + l;                       // A-matrix row for this lane
    float s1v  = s1[row];
    float mv   = rmax[row];
    float rinv = 1.f / rsum[row];

    v8f acc[NT];
    v8f z = {};
#pragma unroll
    for (int t = 0; t < NT; ++t) acc[t] = z;

    // ---- async stage of one 32-wide K chunk of WhT into LDS buffer `buf` ----
    auto stage = [&](int buf, int k0) {
        // D*4 chunks of 16B; thread -> (d = idx>>2, seg = idx&3)
#pragma unroll
        for (int idx = tid; idx < D * 4; idx += 256) {
            int d = idx >> 2, seg = idx & 3;
            unsigned goff = (unsigned)(((size_t)d * NN + k0 + seg * 8) * 2);
            unsigned loff =
                (unsigned)(size_t)&smem[((buf * D + d) * 32) + seg * 8];
            async_g2lds_b128(loff, goff, (const void*)WhT);
        }
    };

    stage(0, 0);
    const unsigned* mrow = mask + (size_t)row * MWORDS;
    for (int k0 = 0; k0 < NN; k0 += 32) {
        int buf = (k0 >> 5) & 1;
        wait_async0();
        __syncthreads();                       // staged chunk visible to all waves
        if (k0 + 32 < NN) stage(buf ^ 1, k0 + 32);

        unsigned wbits = mrow[k0 >> 5];
        v16h a;
#pragma unroll
        for (int v = 0; v < 8; ++v) {
            int K = 2 * v + (half ? 16 : 0);
            float2 sv = *(const float2*)(s2 + k0 + K);
            float e0 = s1v + sv.x; e0 = e0 > 0.f ? e0 : SLOPE * e0;
            float e1 = s1v + sv.y; e1 = e1 > 0.f ? e1 : SLOPE * e1;
            float p0 = ((wbits >> K) & 1u)       ? __expf(e0 - mv) * rinv : 0.f;
            float p1 = ((wbits >> (K + 1)) & 1u) ? __expf(e1 - mv) * rinv : 0.f;
            a[2 * v]     = (_Float16)p0;
            a[2 * v + 1] = (_Float16)p1;
        }
        int klo = half ? 16 : 0;               // within-chunk K offset
#pragma unroll
        for (int t = 0; t < NT; ++t) {
            v16h b = *(const v16h*)&smem[((buf * D + t * 16 + l) * 32) + klo];
            acc[t] = __builtin_amdgcn_wmma_f32_16x16x32_f16(false, a, false, b,
                                                            (short)0, acc[t],
                                                            false, false);
        }
        __syncthreads();                       // all reads done before re-stage
    }

    int colBase = h * D;                        // head concat offset
#pragma unroll
    for (int t = 0; t < NT; ++t) {
#pragma unroll
        for (int r = 0; r < 8; ++r) {
            int orow = m0 + r + (half ? 8 : 0);
            int ocol = t * 16 + l;
            float wv  = Wh[(size_t)orow * D + ocol];
            float res = KEEP * acc[t][r] + (1.f - KEEP) * wv;
            if (ELU) res = res > 0.f ? res : (__expf(res) - 1.f);
            out[(size_t)orow * outLd + colBase + ocol] = res;
        }
    }
}

// ---------------------------------------------------------------------------
// Host orchestration
// ---------------------------------------------------------------------------
extern "C" void kernel_launch(void* const* d_in, const int* in_sizes, int n_in,
                              void* d_out, int out_size, void* d_ws, size_t ws_size,
                              hipStream_t stream) {
    const float* features = (const float*)d_in[0];          // [4096,128]
    const float* adj      = (const float*)d_in[1];          // [2,4096,4096]
    const float* W_heads  = (const float*)d_in[2];          // [2,4,128,64]
    const float* a1_heads = (const float*)d_in[3];          // [2,4,64]
    const float* a2_heads = (const float*)d_in[4];          // [2,4,64]
    const float* W_out    = (const float*)d_in[5];          // [2,256,128]
    const float* a1_out   = (const float*)d_in[6];          // [2,128]
    const float* a2_out   = (const float*)d_in[7];          // [2,128]
    float* outp = (float*)d_out;                            // [4096,128]

    char* bump = (char*)d_ws;
    auto alloc = [&](size_t bytes) {
        void* r = (void*)bump;
        bump += (bytes + 255) & ~(size_t)255;
        return r;
    };
    unsigned*  maskb   = (unsigned*)alloc((size_t)NN * MWORDS * 4);      // 2 MB
    float*     Wh_all  = (float*)alloc((size_t)4 * NN * 64 * 4);         // 4 MB
    _Float16*  WhT_all = (_Float16*)alloc((size_t)4 * NN * 64 * 2);      // 2 MB
    float*     s1h     = (float*)alloc((size_t)4 * NN * 4);
    float*     s2h     = (float*)alloc((size_t)4 * NN * 4);
    float*     rmaxh   = (float*)alloc((size_t)4 * NN * 4);
    float*     rsumh   = (float*)alloc((size_t)4 * NN * 4);
    float*     xbuf    = (float*)alloc((size_t)NN * 256 * 4);            // 4 MB
    float*     WhO     = (float*)alloc((size_t)NN * 128 * 4);            // 2 MB
    _Float16*  WhOT    = (_Float16*)alloc((size_t)NN * 128 * 2);         // 1 MB
    float*     s1o     = (float*)alloc((size_t)NN * 4);
    float*     s2o     = (float*)alloc((size_t)NN * 4);
    float*     rmaxo   = (float*)alloc((size_t)NN * 4);
    float*     rsumo   = (float*)alloc((size_t)NN * 4);
    float*     feat1   = (float*)alloc((size_t)NN * 128 * 4);            // 2 MB

    const float* cur = features;
    for (int i = 0; i < 2; ++i) {
        // mask bits for this layer (reads adj once)
        build_mask<<<(NN * MWORDS) / 256, 256, 0, stream>>>(
            adj + (size_t)i * NN * NN, maskb);

        // ---- heads: Wh = cur @ W_heads[i][h]  (4 batched) ----
        gemm_wmma<<<dim3(NN / 16, 64 / 16, 4), 32, 0, stream>>>(
            cur, W_heads + (size_t)i * 4 * 128 * 64, Wh_all,
            /*K=*/128, /*N=*/64, /*bZ=*/128 * 64, /*cZ=*/(long)NN * 64);
        rowdot2<<<dim3(NN, 4), 32, 0, stream>>>(
            Wh_all, a1_heads + (size_t)i * 4 * 64, a2_heads + (size_t)i * 4 * 64,
            s1h, s2h, /*D=*/64, /*whZ=*/(long)NN * 64, /*aS=*/64, /*sS=*/NN);
        cvt_t<<<dim3((NN * 64 + 255) / 256, 4), 256, 0, stream>>>(
            Wh_all, WhT_all, 64, (long)NN * 64, (long)NN * 64);
        softmax_stats<4><<<NN / 8, 256, 0, stream>>>(maskb, s1h, s2h, rmaxh, rsumh, NN);
        attn_apply<4, false><<<dim3(NN / 128, 4), 256, 0, stream>>>(
            maskb, s1h, s2h, rmaxh, rsumh, WhT_all, Wh_all, xbuf,
            /*outLd=*/256, /*sS=*/NN, (long)NN * 64, (long)NN * 64);

        // ---- out stage: WhO = xbuf @ W_out[i] ----
        gemm_wmma<<<dim3(NN / 16, 128 / 16, 1), 32, 0, stream>>>(
            xbuf, W_out + (size_t)i * 256 * 128, WhO, /*K=*/256, /*N=*/128, 0, 0);
        rowdot2<<<dim3(NN, 1), 32, 0, stream>>>(
            WhO, a1_out + (size_t)i * 128, a2_out + (size_t)i * 128,
            s1o, s2o, 128, 0, 0, 0);
        cvt_t<<<dim3((NN * 128 + 255) / 256, 1), 256, 0, stream>>>(
            WhO, WhOT, 128, 0, 0);
        softmax_stats<1><<<NN / 8, 256, 0, stream>>>(maskb, s1o, s2o, rmaxo, rsumo, NN);

        float* dst = (i == 1) ? outp : feat1;
        attn_apply<8, true><<<dim3(NN / 128, 1), 256, 0, stream>>>(
            maskb, s1o, s2o, rmaxo, rsumo, WhOT, WhO, dst,
            /*outLd=*/128, /*sS=*/NN, 0, 0);
        cur = feat1;
    }
    (void)in_sizes; (void)n_in; (void)out_size; (void)ws_size;
}
